// CepstralBlock_69750268887568
// MI455X (gfx1250) — matrix operand order repaired and checked by reading, not compiled, not run on hardware
//
#include <hip/hip_runtime.h>

typedef __attribute__((ext_vector_type(16))) __bf16 v16bf;
typedef __attribute__((ext_vector_type(8)))  float  v8f;
typedef __attribute__((ext_vector_type(4)))  float  v4f;
typedef __attribute__((ext_vector_type(8)))  unsigned short u16x8;
typedef __attribute__((ext_vector_type(4)))  unsigned short u16x4;

#define D_DIM 16
#define HW 4096          // H*W = 64*64 (row stride between adjacent d)
#define C_DIM 128
#define K_TAPS 4
#define WSTR 136         // padded LDS stride (halves) for W^T rows (16B aligned)
#define YSTR 136         // padded LDS stride (halves) for y rows
#define WAVES 4
#define ROWS_PER_WAVE 16
#define ROWS_PER_WG (WAVES * ROWS_PER_WAVE)

__device__ __forceinline__ unsigned short f2bf(float f) {
  // round-to-nearest-even float -> bf16
  unsigned int u = __float_as_uint(f);
  u += 0x7FFFu + ((u >> 16) & 1u);
  return (unsigned short)(u >> 16);
}

__global__ __launch_bounds__(128)
void cepstral_gemm_kernel(const float* __restrict__ x,
                          const float* __restrict__ kern,
                          const float* __restrict__ W,
                          const float* __restrict__ bias,
                          float* __restrict__ out) {
  __shared__ unsigned short wt_lds[C_DIM * WSTR];               // W^T bf16: [o][c]
  __shared__ unsigned short y_lds[WAVES * ROWS_PER_WAVE * YSTR];// per-wave y tiles
  __shared__ float kern_lds[16 * K_TAPS];

  const int tid  = threadIdx.x;
  const int lane = tid & 31;
  const int wave = tid >> 5;

  // ---- stage W^T (bf16) into LDS; thread t owns output column o = t ----
  {
    const int o = tid;  // 0..127, global reads coalesced across threads
    #pragma unroll 4
    for (int c = 0; c < C_DIM; ++c)
      wt_lds[o * WSTR + c] = f2bf(W[c * C_DIM + o]);
  }
  if (tid < 16 * K_TAPS) kern_lds[tid] = kern[tid];
  __syncthreads();

  // ---- causal depthwise conv along D: build this wave's 16-row y tile ----
  const int row_base = blockIdx.x * ROWS_PER_WG + wave * ROWS_PER_WAVE;
  const int c0 = lane * 4;                       // 4 contiguous channels/lane
  v4f wv[K_TAPS];
  {
    const int cg = (lane & 3) * 4;               // c0 % 16 (kernel tiles over C)
    #pragma unroll
    for (int k = 0; k < K_TAPS; ++k)
      #pragma unroll
      for (int j = 0; j < 4; ++j)
        wv[k][j] = kern_lds[(cg + j) * K_TAPS + k];
  }
  unsigned short* myy = &y_lds[wave * ROWS_PER_WAVE * YSTR];
  #pragma unroll
  for (int m = 0; m < ROWS_PER_WAVE; ++m) {
    const int r = row_base + m;                  // flat (b,d,h,w) row index
    const int d = (r >> 12) & (D_DIM - 1);       // depth coordinate
    v4f acc = {0.f, 0.f, 0.f, 0.f};
    #pragma unroll
    for (int k = 0; k < K_TAPS; ++k) {
      if (d >= k) {                              // causal zero padding
        const v4f xv = *(const v4f*)(x + (size_t)(r - k * HW) * C_DIM + c0);
        acc += xv * wv[k];
      }
    }
    u16x4 p;
    #pragma unroll
    for (int j = 0; j < 4; ++j) p[j] = f2bf(acc[j]);
    *(u16x4*)&myy[m * YSTR + c0] = p;            // ds_store_b64
  }
  __syncthreads();

  // ---- [16x128] @ [128x128] tile GEMM via v_wmma_f32_16x16x32_bf16 ----
  const int hi   = (lane >= 16) ? 1 : 0;
  const int mrow = lane & 15;

  // A fragments: 16x32 bf16 per K-block.
  // lanes 0-15: halves 0..7 -> K 0..7,  halves 8..15 -> K 16..23
  // lanes 16-31: halves 0..7 -> K 8..15, halves 8..15 -> K 24..31
  v16bf a[4];
  #pragma unroll
  for (int kb = 0; kb < 4; ++kb) {
    const int k0 = kb * 32 + hi * 8;
    union { u16x8 u[2]; v16bf v; } cv;
    cv.u[0] = *(const u16x8*)&myy[mrow * YSTR + k0];        // ds_load_b128
    cv.u[1] = *(const u16x8*)&myy[mrow * YSTR + k0 + 16];
    a[kb] = cv.v;
  }

  #pragma unroll
  for (int nt = 0; nt < 8; ++nt) {
    const int o = nt * 16 + mrow;               // this lane's output column
    const float bv = bias[o];
    v8f cacc = {bv, bv, bv, bv, bv, bv, bv, bv};
    #pragma unroll
    for (int kb = 0; kb < 4; ++kb) {
      // B fragment: 32x16 bf16; lanes 0-15 hold K kb*32+0..7,16..23,
      // lanes 16-31 hold K kb*32+8..15,24..31 (N = lane%16, K contiguous in W^T row)
      const int k0 = kb * 32 + hi * 8;
      union { u16x8 u[2]; v16bf v; } cv;
      cv.u[0] = *(const u16x8*)&wt_lds[o * WSTR + k0];
      cv.u[1] = *(const u16x8*)&wt_lds[o * WSTR + k0 + 16];
      cacc = __builtin_amdgcn_wmma_f32_16x16x32_bf16(
          false, a[kb], false, cv.v, (short)0, cacc, false, false);
    }
    // C/D layout: VGPR i -> M=i (lanes 0-15), M=8+i (lanes 16-31); N = lane%16
    const int mb = row_base + hi * 8;
    #pragma unroll
    for (int i = 0; i < 8; ++i)
      out[(size_t)(mb + i) * C_DIM + o] = cacc[i];
  }
}

extern "C" void kernel_launch(void* const* d_in, const int* in_sizes, int n_in,
                              void* d_out, int out_size, void* d_ws, size_t ws_size,
                              hipStream_t stream) {
  const float* x    = (const float*)d_in[0];  // [4,16,64,64,128] f32
  const float* kern = (const float*)d_in[1];  // [16,4] f32
  const float* W    = (const float*)d_in[2];  // [128,128] f32
  const float* b    = (const float*)d_in[3];  // [128] f32
  float* out = (float*)d_out;                 // [4,16,64,64,128] f32

  const int total_rows = in_sizes[0] / C_DIM;        // 262144 spatial positions
  const int grid = total_rows / ROWS_PER_WG;         // 4096 workgroups
  cepstral_gemm_kernel<<<grid, 128, 0, stream>>>(x, kern, W, b, out);
}
